// CausalSelfAttention_55645596287376
// MI455X (gfx1250) — matrix operand order repaired and checked by reading, not compiled, not run on hardware
//
#include <hip/hip_runtime.h>
#include <hip/hip_bf16.h>

// ---------------------------------------------------------------------------
// CausalSelfAttention forward for MI455X (gfx1250, wave32, WMMA).
// fp32 in -> bf16 WMMA (f32 accum) -> fp32 out.
// ---------------------------------------------------------------------------

typedef __attribute__((ext_vector_type(16))) __bf16 v16bf;
typedef __attribute__((ext_vector_type(8)))  __bf16 v8bf;
typedef __attribute__((ext_vector_type(8)))  float  v8f;

namespace cfg {
constexpr int B  = 2;
constexpr int T  = 2048;
constexpr int D  = 2048;
constexpr int H  = 16;
constexpr int Hd = 128;           // D / H
constexpr int D3 = 3 * D;
constexpr int BT = B * T;         // 4096 rows
}

__device__ __forceinline__ v16bf pack16(v8bf lo, v8bf hi) {
  v16bf r;
#pragma unroll
  for (int i = 0; i < 8; ++i) { r[i] = lo[i]; r[i + 8] = hi[i]; }
  return r;
}

__device__ __forceinline__ v8f wmma_bf16(v16bf a, v16bf b, v8f c) {
  // D = A(16x32) * B(32x16) + C, f32 accumulate.
  return __builtin_amdgcn_wmma_f32_16x16x32_bf16(
      /*neg_a=*/false, a, /*neg_b=*/false, b,
      /*c_mod=*/(short)0, c, /*reuse_a=*/false, /*reuse_b=*/false);
}

// ---------------------------------------------------------------------------
// fp32 -> bf16 straight convert
// ---------------------------------------------------------------------------
__global__ void f32_to_bf16_kernel(const float* __restrict__ in,
                                   __bf16* __restrict__ out, int n) {
  int i = blockIdx.x * blockDim.x + threadIdx.x;
  if (i < n) out[i] = (__bf16)in[i];
}

// W is K x N row-major; write Wt as N x K (so WMMA B-fragments load contiguous K).
__global__ void transpose_f32_to_bf16_kernel(const float* __restrict__ W,
                                             __bf16* __restrict__ Wt,
                                             int K, int N) {
  int i = blockIdx.x * blockDim.x + threadIdx.x;
  if (i < K * N) {
    int k = i / N;
    int n = i - k * N;
    Wt[(size_t)n * K + k] = (__bf16)W[i];
  }
}

// ---------------------------------------------------------------------------
// bf16 WMMA GEMM:  C[M,N] (f32) = A[M,K] (bf16 row-major) * Bt[N,K]^T (bf16)
// One wave computes a 16x64 tile (A fragment reused across 4 accumulators).
// ---------------------------------------------------------------------------
__global__ void gemm_bf16_wmma_kernel(const __bf16* __restrict__ A,
                                      const __bf16* __restrict__ Bt,
                                      float* __restrict__ C,
                                      int M, int N, int K) {
  const int wave = (int)((blockIdx.x * blockDim.x + threadIdx.x) >> 5);
  const int lane = threadIdx.x & 31;
  const int half = lane >> 4;     // K-phase select within fragment
  const int ln   = lane & 15;

  const int nTiles = N >> 6;                 // tiles of 64 columns
  const int mt = wave / nTiles;
  const int nt = wave - mt * nTiles;
  if (mt >= (M >> 4)) return;                // wave-uniform, EXEC stays all-1s

  const size_t m0 = (size_t)mt * 16;
  const size_t n0 = (size_t)nt * 64;

  // A: lane = row, 16 contiguous K (offset 16*half)
  const __bf16* arow = A + (m0 + ln) * K + half * 16;
  // B: lane = column, 16 contiguous K (offset 16*half)
  const __bf16* b0r = Bt + (n0 +  0 + ln) * (size_t)K + half * 16;
  const __bf16* b1r = Bt + (n0 + 16 + ln) * (size_t)K + half * 16;
  const __bf16* b2r = Bt + (n0 + 32 + ln) * (size_t)K + half * 16;
  const __bf16* b3r = Bt + (n0 + 48 + ln) * (size_t)K + half * 16;

  v8f acc0 = {}, acc1 = {}, acc2 = {}, acc3 = {};

  for (int k = 0; k < K; k += 32) {
    __builtin_prefetch(arow + k + 256, 0, 0);   // global_prefetch_b8
    __builtin_prefetch(b0r  + k + 256, 0, 0);

    v16bf a  = pack16(*(const v8bf*)(arow + k), *(const v8bf*)(arow + k + 8));
    v16bf b0 = pack16(*(const v8bf*)(b0r + k),  *(const v8bf*)(b0r + k + 8));
    v16bf b1 = pack16(*(const v8bf*)(b1r + k),  *(const v8bf*)(b1r + k + 8));
    v16bf b2 = pack16(*(const v8bf*)(b2r + k),  *(const v8bf*)(b2r + k + 8));
    v16bf b3 = pack16(*(const v8bf*)(b3r + k),  *(const v8bf*)(b3r + k + 8));

    acc0 = wmma_bf16(a, b0, acc0);
    acc1 = wmma_bf16(a, b1, acc1);
    acc2 = wmma_bf16(a, b2, acc2);
    acc3 = wmma_bf16(a, b3, acc3);
  }

  // C/D layout: row = r + 8*half, col = ln (+16j per accumulator)
  float* crow = C + (m0 + half * 8) * N + n0 + ln;
#pragma unroll
  for (int r = 0; r < 8; ++r) {
    crow[(size_t)r * N +  0] = acc0[r];
    crow[(size_t)r * N + 16] = acc1[r];
    crow[(size_t)r * N + 32] = acc2[r];
    crow[(size_t)r * N + 48] = acc3[r];
  }
}

// ---------------------------------------------------------------------------
// RoPE on Q,K + head split + bf16 convert.  Q gets the 1/sqrt(Hd) score scale
// folded in.  V is written transposed (B,H,Hd,T) so attention B-fragments
// load contiguous kv.
// One thread per (b,t,h,j) rotation pair (j, j+64).
// ---------------------------------------------------------------------------
__global__ void rope_split_kernel(const float* __restrict__ qkv,
                                  __bf16* __restrict__ Qh,
                                  __bf16* __restrict__ Kh,
                                  __bf16* __restrict__ Vt) {
  using namespace cfg;
  const int tid = blockIdx.x * blockDim.x + threadIdx.x;  // B*T*H*64
  const int j = tid & 63;
  const int h = (tid >> 6) & (H - 1);
  const int t = (tid >> 10) & (T - 1);
  const int b = tid >> 21;

  const float* base = qkv + ((size_t)b * T + t) * D3 + h * Hd;
  const float q1 = base[j],         q2 = base[j + 64];
  const float k1 = base[D + j],     k2 = base[D + j + 64];
  const float v1 = base[2 * D + j], v2 = base[2 * D + j + 64];

  // inv_freq = 10000^(-j/64)
  const float invf  = __expf(-(float)j * (__logf(10000.0f) / 64.0f));
  const float theta = (float)t * invf;
  const float c = __cosf(theta), s = __sinf(theta);
  const float qs = 0.08838834764831845f;   // Hd^-0.5

  const size_t qoff = ((size_t)(b * H + h) * T + t) * Hd;
  Qh[qoff + j]      = (__bf16)((q1 * c - q2 * s) * qs);
  Qh[qoff + j + 64] = (__bf16)((q2 * c + q1 * s) * qs);
  Kh[qoff + j]      = (__bf16)(k1 * c - k2 * s);
  Kh[qoff + j + 64] = (__bf16)(k2 * c + k1 * s);

  const size_t voff = (size_t)(b * H + h) * Hd * T;
  Vt[voff + (size_t)j * T + t]        = (__bf16)v1;
  Vt[voff + (size_t)(j + 64) * T + t] = (__bf16)v2;
}

// ---------------------------------------------------------------------------
// Flash attention: one wave owns (b, h, 16-row q tile); iterates kv in steps
// of 32 with online softmax.  S = Q*K^T via 8 WMMA, O += P*V via 8 WMMA.
// P crosses C-layout -> A-layout through a per-wave LDS tile (s_wait_dscnt).
// Output written bf16 into (B,T,D) layout for the final projection GEMM.
// ---------------------------------------------------------------------------
#define FA_WAVES 4

__global__ void flash_attn_kernel(const __bf16* __restrict__ Qh,
                                  const __bf16* __restrict__ Kh,
                                  const __bf16* __restrict__ Vt,
                                  __bf16* __restrict__ AttnH) {
  using namespace cfg;
  __shared__ __align__(16) __bf16 plds[FA_WAVES][16 * 32];

  const int wv   = threadIdx.x >> 5;
  const int lane = threadIdx.x & 31;
  const int half = lane >> 4;
  const int ln   = lane & 15;

  const int w  = blockIdx.x * FA_WAVES + wv;
  const int qt = w & (T / 16 - 1);       // 7 bits
  const int h  = (w >> 7) & (H - 1);     // 4 bits
  const int b  = w >> 11;
  const int qbase = qt * 16;

  // Q A-fragments: 16x128 split into 4 K=32 fragments, loaded once.
  const __bf16* qp = Qh + ((size_t)(b * H + h) * T + qbase + ln) * Hd;
  v16bf qa[4];
#pragma unroll
  for (int c = 0; c < 4; ++c) {
    const __bf16* p = qp + c * 32 + half * 16;
    qa[c] = pack16(*(const v8bf*)p, *(const v8bf*)(p + 8));
  }

  v8f o[8] = {};                 // 16x128 f32 output accumulator
  float m_run[8], l_run[8];
#pragma unroll
  for (int r = 0; r < 8; ++r) { m_run[r] = -3.0e38f; l_run[r] = 0.0f; }

  const __bf16* kbase = Kh + (size_t)(b * H + h) * T * Hd;
  const __bf16* vbase = Vt + (size_t)(b * H + h) * Hd * T;
  __bf16* pl = &plds[wv][0];

  for (int kv = 0; kv < qbase + 16; kv += 32) {
    // ---- S = Q * K^T for 32 kv positions (two 16x16 tiles) ----
    v8f s0 = {}, s1 = {};
#pragma unroll
    for (int c = 0; c < 4; ++c) {
      const int kb = c * 32 + half * 16;
      const __bf16* k0 = kbase + (size_t)(kv + ln) * Hd + kb;
      const __bf16* k1 = kbase + (size_t)(kv + 16 + ln) * Hd + kb;
      v16bf b0 = pack16(*(const v8bf*)k0, *(const v8bf*)(k0 + 8));
      v16bf b1 = pack16(*(const v8bf*)k1, *(const v8bf*)(k1 + 8));
      s0 = wmma_bf16(qa[c], b0, s0);
      s1 = wmma_bf16(qa[c], b1, s1);
    }

    // ---- causal mask + online softmax (row-wise across 16-lane groups) ----
    float corr[8];
#pragma unroll
    for (int r = 0; r < 8; ++r) {
      const int row = qbase + r + 8 * half;
      float v0 = (kv + ln      <= row) ? s0[r] : -3.0e38f;
      float v1 = (kv + 16 + ln <= row) ? s1[r] : -3.0e38f;

      float tm = fmaxf(v0, v1);
      tm = fmaxf(tm, __shfl_xor(tm, 1));
      tm = fmaxf(tm, __shfl_xor(tm, 2));
      tm = fmaxf(tm, __shfl_xor(tm, 4));
      tm = fmaxf(tm, __shfl_xor(tm, 8));

      const float mnew = fmaxf(m_run[r], tm);
      const float p0 = __expf(v0 - mnew);
      const float p1 = __expf(v1 - mnew);
      float ps = p0 + p1;
      ps += __shfl_xor(ps, 1);
      ps += __shfl_xor(ps, 2);
      ps += __shfl_xor(ps, 4);
      ps += __shfl_xor(ps, 8);

      const float cr = __expf(m_run[r] - mnew);
      l_run[r] = l_run[r] * cr + ps;
      m_run[r] = mnew;
      corr[r]  = cr;

      const int rowl = r + 8 * half;
      pl[rowl * 32 + ln]      = (__bf16)p0;   // P tile, C-layout -> LDS row-major
      pl[rowl * 32 + 16 + ln] = (__bf16)p1;
    }

    // rescale running output by exp(m_old - m_new)
#pragma unroll
    for (int a = 0; a < 8; ++a)
#pragma unroll
      for (int r = 0; r < 8; ++r) o[a][r] *= corr[r];

    // all lanes' P stores must land before cross-lane A-fragment reload
    asm volatile("s_wait_dscnt 0" ::: "memory");

    const __bf16* pp = pl + ln * 32 + half * 16;
    v16bf pa = pack16(*(const v8bf*)pp, *(const v8bf*)(pp + 8));

    // ---- O += P(16x32) * V(32x128) ----
#pragma unroll
    for (int a = 0; a < 8; ++a) {
      const __bf16* vp = vbase + (size_t)(a * 16 + ln) * T + kv + half * 16;
      v16bf vb = pack16(*(const v8bf*)vp, *(const v8bf*)(vp + 8));
      o[a] = wmma_bf16(pa, vb, o[a]);
    }
  }

  // ---- normalize and write to (B,T,D) bf16 for the projection GEMM ----
#pragma unroll
  for (int r = 0; r < 8; ++r) {
    const float inv = 1.0f / l_run[r];
    const int t = qbase + r + 8 * half;
    const size_t rowoff = ((size_t)b * T + t) * D + h * Hd;
#pragma unroll
    for (int a = 0; a < 8; ++a)
      AttnH[rowoff + a * 16 + ln] = (__bf16)(o[a][r] * inv);
  }
}

// ---------------------------------------------------------------------------
// Host-side orchestration
// ---------------------------------------------------------------------------
extern "C" void kernel_launch(void* const* d_in, const int* in_sizes, int n_in,
                              void* d_out, int out_size, void* d_ws, size_t ws_size,
                              hipStream_t stream) {
  using namespace cfg;
  (void)in_sizes; (void)n_in; (void)out_size; (void)ws_size;

  const float* x     = (const float*)d_in[0];   // (B,T,D)
  const float* w_qkv = (const float*)d_in[1];   // (D, 3D)
  const float* w_out = (const float*)d_in[2];   // (D, D)
  float* out = (float*)d_out;                   // (B,T,D) f32

  char* ws = (char*)d_ws;
  auto take = [&](size_t bytes) {
    char* p = ws;
    ws += (bytes + 255) & ~(size_t)255;
    return p;
  };

  __bf16* Xh     = (__bf16*)take((size_t)BT * D * 2);        // x in bf16
  __bf16* Wqkv_t = (__bf16*)take((size_t)D3 * D * 2);        // w_qkv^T bf16
  __bf16* Wo_t   = (__bf16*)take((size_t)D  * D * 2);        // w_out^T bf16
  float*  QKVf   = (float*) take((size_t)BT * D3 * 4);       // qkv f32
  __bf16* Qhb    = (__bf16*)take((size_t)B * H * T * Hd * 2);
  __bf16* Khb    = (__bf16*)take((size_t)B * H * T * Hd * 2);
  __bf16* Vtb    = (__bf16*)take((size_t)B * H * Hd * T * 2);
  __bf16* AttnH  = (__bf16*)take((size_t)BT * D * 2);

  // 1) converts / weight transposes
  f32_to_bf16_kernel<<<(BT * D) / 256, 256, 0, stream>>>(x, Xh, BT * D);
  transpose_f32_to_bf16_kernel<<<(D * D3) / 256, 256, 0, stream>>>(w_qkv, Wqkv_t, D, D3);
  transpose_f32_to_bf16_kernel<<<(D * D) / 256, 256, 0, stream>>>(w_out, Wo_t, D, D);

  // 2) QKV projection: (4096 x 2048) * (2048 x 6144)
  {
    int waves = (BT / 16) * (D3 / 64);
    gemm_bf16_wmma_kernel<<<waves / 4, 128, 0, stream>>>(Xh, Wqkv_t, QKVf, BT, D3, D);
  }

  // 3) RoPE + head split (+ fold Hd^-0.5 into Q, transpose V)
  rope_split_kernel<<<(B * T * H * 64) / 256, 256, 0, stream>>>(QKVf, Qhb, Khb, Vtb);

  // 4) causal flash attention
  {
    int waves = B * H * (T / 16);
    flash_attn_kernel<<<waves / FA_WAVES, 32 * FA_WAVES, 0, stream>>>(Qhb, Khb, Vtb, AttnH);
  }

  // 5) output projection: (4096 x 2048) * (2048 x 2048) -> f32 d_out
  {
    int waves = (BT / 16) * (D / 64);
    gemm_bf16_wmma_kernel<<<waves / 4, 128, 0, stream>>>(AttnH, Wo_t, out, BT, D, D);
  }
}
